// GATPerception_22539988370022
// MI455X (gfx1250) — matrix operand order repaired and checked by reading, not compiled, not run on hardware
//
#include <hip/hip_runtime.h>
#include <hip/hip_bf16.h>
#include <math.h>

// ---------------------------------------------------------------------------
// GATPerception for MI455X (gfx1250, wave32, WMMA).
//
// All large GEMMs run through v_wmma_f32_16x16x32_f16 (f16 operands, fp32
// accum).  One wave computes a 16x64 output strip: one A fragment feeds four
// WMMAs (4x A reuse).  The GAT Wh projection stores transposed (WhT[d][j]) so
// alpha@Wh also uses the contiguous-B fast path -- there is no strided-B GEMM
// path anywhere.
//
// Workspace requirement: ~110 MB.
//
// d_in order assumption: setup_inputs() dict order, nested `params` flattened
// jax-tree style (dict keys sorted):
//   0 global_memory  1 env_global_node  2 curr_embedding  3 goal_embedding
//   4 step  5 global_time  6 global_mask  7 global_A
//   8..23  curr_dec: b1,b2,be1,be2,bk,bo,bq,bv,g1,g2,w1,w2,wk,wo,wq,wv
//   24 fe_b1  25 fe_b2  26 fe_w1  27 fe_w2
//   28..30 gat[0]: a_dst,a_src,w    31..33 gat[1]: a_dst,a_src,w
//   34..49 goal_dec (same order as curr_dec)
// d_out: [curr_out (16x512) | goal_ctx (16x512) | new_env (16x512)]  fp32
// ---------------------------------------------------------------------------

constexpr int B   = 16;
constexpr int N   = 1024;
constexpr int D   = 512;
constexpr int DFF = 2048;
constexpr int H   = 8;
constexpr int DH  = 64;          // D / H
constexpr int L   = N + 2;       // 1026 valid sequence positions
constexpr int Lp  = 1056;        // padded to multiple of 32 (66 * 16)
constexpr float LOG1E4 = 9.210340371976184f;   // ln(10000)

typedef __attribute__((ext_vector_type(16))) _Float16 v16h;
typedef __attribute__((ext_vector_type(8)))  _Float16 v8h;
typedef __attribute__((ext_vector_type(8)))  float    v8f;

enum { ACT_NONE = 0, ACT_RELU = 1, ACT_ELU = 2 };

__device__ __forceinline__ v16h load_b_frag(const _Float16* p) {
  // B fragment (32x16, 16-bit): lane half g holds K = g*16 + {0..15},
  // contiguous 32 bytes (caller bakes the +g*16 into p).
  v8h lo = *reinterpret_cast<const v8h*>(p);
  v8h hi = *reinterpret_cast<const v8h*>(p + 8);
  v16h r;
#pragma unroll
  for (int i = 0; i < 8; ++i) { r[i] = lo[i]; r[8 + i] = hi[i]; }
  return r;
}

// ---------------------------------------------------------------------------
// WMMA GEMM:  C = act( A[M,K] @ W[Ncols,K]^T + bias )
//   CT=false: C[r*ldc + c]   (row-major [M,Ncols])
//   CT=true : C[c*ldc + r]   (transposed store, ldc = M-dimension stride)
// One wave -> 16 rows x 64 cols (4 accumulators, 1 A frag reused 4x).
// 8 waves/block stacked along M.  grid = (ceil(M/128), Ncols/64, batches).
// Requires: M mult of 16, Ncols mult of 64, K mult of 32, lda/ldb mult of 8,
// 16B-aligned bases (guaranteed by 256B-aligned workspace allocator).
// ---------------------------------------------------------------------------
template <bool CT>
__global__ __launch_bounds__(256) void wmma_gemm_kernel(
    const _Float16* __restrict__ A, long long strideA, int lda,
    const _Float16* __restrict__ Bm, long long strideB, int ldb,
    const float* __restrict__ bias,
    _Float16* __restrict__ C, long long strideC, int ldc,
    int M, int K, int act)
{
  const int wave = threadIdx.x >> 5;
  const int lane = threadIdx.x & 31;
  const int mt = blockIdx.x * 8 + wave;
  if (mt * 16 >= M) return;                 // wave-uniform: EXEC stays all-1s
  const int g   = lane >> 4;                // lane half (0/1)
  const int l16 = lane & 15;
  const int nc0 = blockIdx.y * 64;

  // A fragment (16x32, 16-bit): lane half g holds K = g*16+{0..15}
  // (contiguous, matching the B fragment).
  const _Float16* Ap = A + (long long)blockIdx.z * strideA
                         + (long long)(mt * 16 + l16) * lda + g * 16;
  const _Float16* Bp = Bm + (long long)blockIdx.z * strideB
                          + (long long)(nc0 + l16) * ldb + g * 16;
  const long long bstep = (long long)16 * ldb;

  v8f acc0 = {}, acc1 = {}, acc2 = {}, acc3 = {};
#pragma unroll 2
  for (int k0 = 0; k0 < K; k0 += 32) {
    v16h a;
    {
      v8h lo = *reinterpret_cast<const v8h*>(Ap);
      v8h hi = *reinterpret_cast<const v8h*>(Ap + 8);
#pragma unroll
      for (int i = 0; i < 8; ++i) { a[i] = lo[i]; a[8 + i] = hi[i]; }
    }
    v16h b0 = load_b_frag(Bp);
    v16h b1 = load_b_frag(Bp + bstep);
    v16h b2 = load_b_frag(Bp + 2 * bstep);
    v16h b3 = load_b_frag(Bp + 3 * bstep);
    acc0 = __builtin_amdgcn_wmma_f32_16x16x32_f16(false, a, false, b0, (short)0, acc0, false, false);
    acc1 = __builtin_amdgcn_wmma_f32_16x16x32_f16(false, a, false, b1, (short)0, acc1, false, false);
    acc2 = __builtin_amdgcn_wmma_f32_16x16x32_f16(false, a, false, b2, (short)0, acc2, false, false);
    acc3 = __builtin_amdgcn_wmma_f32_16x16x32_f16(false, a, false, b3, (short)0, acc3, false, false);
    Ap += 32;
    Bp += 32;
    if (k0 + 32 < K) __builtin_prefetch(Ap, 0, 1);   // global_prefetch_b8
  }

  _Float16* Cb = C + (long long)blockIdx.z * strideC;
  // C/D layout: VGPR v holds M = mt*16 + g*8 + v, N = tile*16 + l16.
#pragma unroll
  for (int v = 0; v < 8; ++v) {
    const int r = mt * 16 + g * 8 + v;
    float vals[4] = { acc0[v], acc1[v], acc2[v], acc3[v] };
#pragma unroll
    for (int t = 0; t < 4; ++t) {
      const int c = nc0 + t * 16 + l16;
      float val = vals[t];
      if (bias) val += bias[c];
      if (act == ACT_RELU)     val = fmaxf(val, 0.0f);
      else if (act == ACT_ELU) val = (val > 0.0f) ? val : (expf(val) - 1.0f);
      if (r < M) {
        if (CT) Cb[(long long)c * ldc + r] = (_Float16)val;
        else    Cb[(long long)r * ldc + c] = (_Float16)val;
      }
    }
  }
}

// ---------------------------------------------------------------------------
__global__ void cvt_f32_to_f16_kernel(const float* __restrict__ s,
                                      _Float16* __restrict__ d, int n)
{
  int i = blockIdx.x * 256 + threadIdx.x;
  if (i < n) d[i] = (_Float16)s[i];
}

// cat[b,n,:] = [ global_memory[b,n,:] + pe(step[b]-time[b,n]) ‖ goal[b,:] ]
__global__ __launch_bounds__(256) void build_cat_kernel(
    const float* __restrict__ gm, const float* __restrict__ goal,
    const int* __restrict__ step, const int* __restrict__ gtime,
    _Float16* __restrict__ cat)
{
  int row = blockIdx.x;            // b*N + n
  int b = row >> 10;
  float pos = (float)(step[b] - gtime[row]);
  for (int d = threadIdx.x; d < D; d += 256) {
    int i = d >> 1;
    float div = expf(-(float)(2 * i) * (LOG1E4 / (float)D));
    float pe = (d & 1) ? cosf(pos * div) : sinf(pos * div);
    cat[(long long)row * (2 * D) + d]     = (_Float16)(gm[(long long)row * D + d] + pe);
    cat[(long long)row * (2 * D) + D + d] = (_Float16)goal[b * D + d];
  }
}

// x[b,0]=env, x[b,1]=curr, rest zero (rows 2..N+1 overwritten by fe GEMM-2)
__global__ void init_x_kernel(const float* __restrict__ env,
                              const float* __restrict__ curr,
                              _Float16* __restrict__ x)
{
  long long idx = (long long)blockIdx.x * 256 + threadIdx.x;
  if (idx >= (long long)B * Lp * D) return;
  int d = (int)(idx % D);
  long long row = idx / D;
  int r = (int)(row % Lp);
  int b = (int)(row / Lp);
  float v = 0.0f;
  if (r == 0)      v = env[b * D + d];
  else if (r == 1) v = curr[b * D + d];
  x[idx] = (_Float16)v;
}

// esrc[b*Lp+i] = sum_d WhT[b][d][i]*a_src[d]  (WhT is [B][D][Lp], coalesced)
__global__ __launch_bounds__(256) void gat_edge_kernel(
    const _Float16* __restrict__ WhT, const float* __restrict__ a_src,
    const float* __restrict__ a_dst, float* __restrict__ esrc,
    float* __restrict__ edst)
{
  int idx = blockIdx.x * 256 + threadIdx.x;      // b*Lp + i
  if (idx >= B * Lp) return;
  int b = idx / Lp;
  int i = idx - b * Lp;
  const _Float16* base = WhT + (long long)b * D * Lp + i;
  float p1 = 0.f, p2 = 0.f;
#pragma unroll 4
  for (int d = 0; d < D; ++d) {
    float w = (float)base[(long long)d * Lp];
    p1 += w * a_src[d];
    p2 += w * a_dst[d];
  }
  esrc[idx] = p1;
  edst[idx] = p2;
}

// alpha[b,i,:] = softmax_j( mask(A_ext, leaky(esrc_i + edst_j)) ), f16 out
__global__ __launch_bounds__(256) void gat_softmax_kernel(
    const float* __restrict__ esrc, const float* __restrict__ edst,
    const float* __restrict__ gA, _Float16* __restrict__ alpha)
{
  __shared__ float sc[Lp];
  __shared__ float red[256];
  int b = blockIdx.x / Lp;
  int i = blockIdx.x % Lp;
  _Float16* arow = alpha + ((long long)b * Lp + i) * Lp;
  if (i >= L) {                                    // pad row -> zeros
    for (int j = threadIdx.x; j < Lp; j += 256) arow[j] = (_Float16)0.0f;
    return;
  }
  float ei = esrc[b * Lp + i];
  float mx = -3.0e38f;
  for (int j = threadIdx.x; j < Lp; j += 256) {
    float s = -1.0e9f;
    if (j < L) {
      float conn;
      if (i < 2 || j < 2 || j == i) conn = 1.0f;   // border rows/cols + eye
      else conn = gA[((long long)b * N + (i - 2)) * N + (j - 2)];
      if (conn > 0.0f) {
        float e = ei + edst[b * Lp + j];
        s = (e > 0.0f) ? e : 0.2f * e;             // LeakyReLU(0.2)
      }
    }
    sc[j] = s;
    mx = fmaxf(mx, s);
  }
  red[threadIdx.x] = mx; __syncthreads();
  for (int s = 128; s > 0; s >>= 1) {
    if (threadIdx.x < s) red[threadIdx.x] = fmaxf(red[threadIdx.x], red[threadIdx.x + s]);
    __syncthreads();
  }
  mx = red[0]; __syncthreads();
  float sum = 0.f;
  for (int j = threadIdx.x; j < Lp; j += 256) {
    float e = expf(sc[j] - mx);
    sc[j] = e; sum += e;
  }
  red[threadIdx.x] = sum; __syncthreads();
  for (int s = 128; s > 0; s >>= 1) {
    if (threadIdx.x < s) red[threadIdx.x] += red[threadIdx.x + s];
    __syncthreads();
  }
  float inv = 1.0f / red[0];
  for (int j = threadIdx.x; j < Lp; j += 256) arow[j] = (_Float16)(sc[j] * inv);
}

// ctx[b, 2+n, :] += pe(step[b]-time[b,n])
__global__ __launch_bounds__(256) void add_pe_kernel(
    _Float16* __restrict__ ctx, const int* __restrict__ step,
    const int* __restrict__ gtime)
{
  int row = blockIdx.x;            // b*N + n
  int b = row >> 10;
  int n = row & 1023;
  float pos = (float)(step[b] - gtime[row]);
  _Float16* c = ctx + ((long long)b * Lp + 2 + n) * D;
  for (int d = threadIdx.x; d < D; d += 256) {
    int i = d >> 1;
    float div = expf(-(float)(2 * i) * (LOG1E4 / (float)D));
    float pe = (d & 1) ? cosf(pos * div) : sinf(pos * div);
    c[d] = (_Float16)((float)c[d] + pe);
  }
}

__global__ void copy_env_kernel(const _Float16* __restrict__ ctx,
                                float* __restrict__ out)
{
  int idx = blockIdx.x * 256 + threadIdx.x;      // B*D
  if (idx >= B * D) return;
  int b = idx / D, d = idx % D;
  out[idx] = (float)ctx[((long long)b * Lp) * D + d];
}

// Single-query multihead attention: out16[b, h*64+d] (one block per (b,h))
__global__ __launch_bounds__(256) void dec_attn_kernel(
    const _Float16* __restrict__ q16, const _Float16* __restrict__ k16,
    const _Float16* __restrict__ v16, const float* __restrict__ mask,
    _Float16* __restrict__ out16)
{
  __shared__ float sc[Lp];
  __shared__ float red[256];
  __shared__ float part[4 * DH];
  int b = blockIdx.x / H;
  int h = blockIdx.x % H;
  const _Float16* qh = q16 + b * D + h * DH;
  for (int j = threadIdx.x; j < Lp; j += 256) {
    float s = -1.0e9f;
    bool valid = (j < L) && (j < 2 || mask[b * N + (j - 2)] > 0.0f);
    if (valid) {
      const _Float16* kh = k16 + ((long long)b * Lp + j) * D + h * DH;
      float acc = 0.f;
#pragma unroll 8
      for (int d = 0; d < DH; ++d) acc += (float)qh[d] * (float)kh[d];
      s = acc * 0.125f;                            // 1/sqrt(64)
    }
    sc[j] = s;
  }
  __syncthreads();
  float mx = -3.0e38f;
  for (int j = threadIdx.x; j < Lp; j += 256) mx = fmaxf(mx, sc[j]);
  red[threadIdx.x] = mx; __syncthreads();
  for (int s = 128; s > 0; s >>= 1) {
    if (threadIdx.x < s) red[threadIdx.x] = fmaxf(red[threadIdx.x], red[threadIdx.x + s]);
    __syncthreads();
  }
  mx = red[0]; __syncthreads();
  float sum = 0.f;
  for (int j = threadIdx.x; j < Lp; j += 256) {
    float e = expf(sc[j] - mx);
    sc[j] = e; sum += e;
  }
  red[threadIdx.x] = sum; __syncthreads();
  for (int s = 128; s > 0; s >>= 1) {
    if (threadIdx.x < s) red[threadIdx.x] += red[threadIdx.x + s];
    __syncthreads();
  }
  float inv = 1.0f / red[0];
  __syncthreads();
  int d = threadIdx.x & (DH - 1);
  int grp = threadIdx.x >> 6;                      // 4 groups
  float p = 0.f;
  for (int j = grp; j < Lp; j += 4)
    p += sc[j] * (float)v16[((long long)b * Lp + j) * D + h * DH + d];
  part[grp * DH + d] = p; __syncthreads();
  if (threadIdx.x < DH) {
    float o = (part[threadIdx.x] + part[DH + threadIdx.x] +
               part[2 * DH + threadIdx.x] + part[3 * DH + threadIdx.x]) * inv;
    out16[b * D + h * DH + threadIdx.x] = (_Float16)o;
  }
}

// out = layernorm(src_f32 + add_f16) * g + be   (optional f32 / f16 outputs)
__global__ __launch_bounds__(256) void add_ln_kernel(
    const float* __restrict__ src, const _Float16* __restrict__ add,
    const float* __restrict__ g, const float* __restrict__ be,
    float* __restrict__ out_f32, _Float16* __restrict__ out_f16)
{
  __shared__ float red[256];
  __shared__ float vals[D];
  int b = blockIdx.x;
  float s = 0.f;
  for (int d = threadIdx.x; d < D; d += 256) {
    float v = src[b * D + d] + (float)add[b * D + d];
    vals[d] = v; s += v;
  }
  red[threadIdx.x] = s; __syncthreads();
  for (int k = 128; k > 0; k >>= 1) {
    if (threadIdx.x < k) red[threadIdx.x] += red[threadIdx.x + k];
    __syncthreads();
  }
  float mean = red[0] / (float)D; __syncthreads();
  float vs = 0.f;
  for (int d = threadIdx.x; d < D; d += 256) {
    float t = vals[d] - mean; vs += t * t;
  }
  red[threadIdx.x] = vs; __syncthreads();
  for (int k = 128; k > 0; k >>= 1) {
    if (threadIdx.x < k) red[threadIdx.x] += red[threadIdx.x + k];
    __syncthreads();
  }
  float rstd = rsqrtf(red[0] / (float)D + 1e-5f);
  for (int d = threadIdx.x; d < D; d += 256) {
    float o = (vals[d] - mean) * rstd * g[d] + be[d];
    if (out_f32) out_f32[b * D + d] = o;
    if (out_f16) out_f16[b * D + d] = (_Float16)o;
  }
}

// ---------------------------------------------------------------------------
static inline void gemmN(hipStream_t st,
                         const _Float16* A, long long sA, int lda,
                         const _Float16* Bm, long long sB, int ldb,
                         const float* bias,
                         _Float16* C, long long sC, int ldc,
                         int M, int Ncols, int K, int act, int batches)
{
  dim3 grid((M + 127) / 128, Ncols / 64, batches);
  wmma_gemm_kernel<false><<<grid, 256, 0, st>>>(A, sA, lda, Bm, sB, ldb,
                                                bias, C, sC, ldc, M, K, act);
}
static inline void gemmT(hipStream_t st,
                         const _Float16* A, long long sA, int lda,
                         const _Float16* Bm, long long sB, int ldb,
                         const float* bias,
                         _Float16* C, long long sC, int ldc,
                         int M, int Ncols, int K, int act, int batches)
{
  dim3 grid((M + 127) / 128, Ncols / 64, batches);
  wmma_gemm_kernel<true><<<grid, 256, 0, st>>>(A, sA, lda, Bm, sB, ldb,
                                               bias, C, sC, ldc, M, K, act);
}

extern "C" void kernel_launch(void* const* d_in, const int* in_sizes, int n_in,
                              void* d_out, int out_size, void* d_ws, size_t ws_size,
                              hipStream_t stream)
{
  (void)in_sizes; (void)n_in; (void)out_size; (void)ws_size;

  const float* global_memory = (const float*)d_in[0];
  const float* env_global    = (const float*)d_in[1];
  const float* curr_emb      = (const float*)d_in[2];
  const float* goal_emb      = (const float*)d_in[3];
  const int*   step          = (const int*)d_in[4];
  const int*   gtime         = (const int*)d_in[5];
  const float* gmask         = (const float*)d_in[6];
  const float* gA            = (const float*)d_in[7];

  struct DecP { const float *b1,*b2,*be1,*be2,*bk,*bo,*bq,*bv,*g1,*g2,*w1,*w2,*wk,*wo,*wq,*wv; };
  auto load_dec = [&](int base) {
    DecP p;
    p.b1=(const float*)d_in[base+0];  p.b2=(const float*)d_in[base+1];
    p.be1=(const float*)d_in[base+2]; p.be2=(const float*)d_in[base+3];
    p.bk=(const float*)d_in[base+4];  p.bo=(const float*)d_in[base+5];
    p.bq=(const float*)d_in[base+6];  p.bv=(const float*)d_in[base+7];
    p.g1=(const float*)d_in[base+8];  p.g2=(const float*)d_in[base+9];
    p.w1=(const float*)d_in[base+10]; p.w2=(const float*)d_in[base+11];
    p.wk=(const float*)d_in[base+12]; p.wo=(const float*)d_in[base+13];
    p.wq=(const float*)d_in[base+14]; p.wv=(const float*)d_in[base+15];
    return p;
  };
  DecP curr_p = load_dec(8);
  const float* fe_b1 = (const float*)d_in[24];
  const float* fe_b2 = (const float*)d_in[25];
  const float* fe_w1 = (const float*)d_in[26];
  const float* fe_w2 = (const float*)d_in[27];
  const float* g0_adst = (const float*)d_in[28];
  const float* g0_asrc = (const float*)d_in[29];
  const float* g0_w    = (const float*)d_in[30];
  const float* g1_adst = (const float*)d_in[31];
  const float* g1_asrc = (const float*)d_in[32];
  const float* g1_w    = (const float*)d_in[33];
  DecP goal_p = load_dec(34);

  // ---- workspace bump allocator (256B aligned) ----
  char* ws = (char*)d_ws;
  size_t off = 0;
  auto alloc16 = [&](size_t e)->_Float16* {
    _Float16* p = (_Float16*)(ws + off); off += ((e * 2 + 255) / 256) * 256; return p; };
  auto allocf = [&](size_t e)->float* {
    float* p = (float*)(ws + off); off += ((e * 4 + 255) / 256) * 256; return p; };

  // f16 weights
  _Float16* w_fe1 = alloc16((size_t)D * 2 * D);
  _Float16* w_fe2 = alloc16((size_t)D * D);
  _Float16* w_g0  = alloc16((size_t)D * D);
  _Float16* w_g1  = alloc16((size_t)D * D);
  struct DecW { _Float16 *wq,*wk,*wv,*wo,*w1,*w2; };
  auto alloc_dec = [&]() {
    DecW w;
    w.wq = alloc16((size_t)D * D);   w.wk = alloc16((size_t)D * D);
    w.wv = alloc16((size_t)D * D);   w.wo = alloc16((size_t)D * D);
    w.w1 = alloc16((size_t)DFF * D); w.w2 = alloc16((size_t)D * DFF);
    return w;
  };
  DecW curr_w = alloc_dec();
  DecW goal_w = alloc_dec();

  // big buffers (reused across phases)
  _Float16* bigA = alloc16((size_t)B * Lp * Lp);  // cat -> alpha -> K
  _Float16* bigB = alloc16((size_t)B * Lp * D);   // h1  -> WhT   -> V
  _Float16* x0   = alloc16((size_t)B * Lp * D);
  _Float16* x1   = alloc16((size_t)B * Lp * D);
  _Float16* goal16 = alloc16((size_t)B * D);
  _Float16* curr16 = alloc16((size_t)B * D);
  _Float16* q16    = alloc16((size_t)B * D);
  _Float16* att16  = alloc16((size_t)B * D);
  _Float16* proj16 = alloc16((size_t)B * D);
  _Float16* ln1_16 = alloc16((size_t)B * D);
  _Float16* ffh    = alloc16((size_t)B * DFF);
  _Float16* ff2    = alloc16((size_t)B * D);
  float* esrc   = allocf((size_t)B * Lp);
  float* edst   = allocf((size_t)B * Lp);
  float* ln1_32 = allocf((size_t)B * D);

  auto cvt = [&](const float* s, _Float16* d, int n) {
    cvt_f32_to_f16_kernel<<<(n + 255) / 256, 256, 0, stream>>>(s, d, n);
  };

  // ---- convert weights ----
  cvt(fe_w1, w_fe1, D * 2 * D);  cvt(fe_w2, w_fe2, D * D);
  cvt(g0_w, w_g0, D * D);        cvt(g1_w, w_g1, D * D);
  auto cvt_dec = [&](const DecP& p, const DecW& w) {
    cvt(p.wq, w.wq, D * D); cvt(p.wk, w.wk, D * D);
    cvt(p.wv, w.wv, D * D); cvt(p.wo, w.wo, D * D);
    cvt(p.w1, w.w1, DFF * D); cvt(p.w2, w.w2, D * DFF);
  };
  cvt_dec(curr_p, curr_w);
  cvt_dec(goal_p, goal_w);
  cvt(goal_emb, goal16, B * D);
  cvt(curr_emb, curr16, B * D);

  // ---- feature encoder ----
  build_cat_kernel<<<B * N, 256, 0, stream>>>(global_memory, goal_emb, step, gtime, bigA);
  // h1 = relu(cat @ fe_w1^T + fe_b1) : M=B*N, K=1024, N=512
  gemmN(stream, bigA, 0, 2 * D, w_fe1, 0, 2 * D, fe_b1,
        bigB, 0, D, B * N, D, 2 * D, ACT_RELU, 1);
  // x0 rows 0/1 = env/curr, pad zero; rows 2.. = h1 @ fe_w2^T + fe_b2
  init_x_kernel<<<(int)(((long long)B * Lp * D + 255) / 256), 256, 0, stream>>>(env_global, curr_emb, x0);
  gemmN(stream, bigB, (long long)N * D, D, w_fe2, 0, D, fe_b2,
        x0 + 2 * D, (long long)Lp * D, D, N, D, D, ACT_NONE, B);

  // ---- GAT layers ----
  auto gat_layer = [&](_Float16* xin, _Float16* xout, _Float16* w,
                       const float* asrc, const float* adst, int act) {
    // WhT[b][d][j] = (x[b] @ w^T)^T  : batched, transposed epilogue store
    gemmT(stream, xin, (long long)Lp * D, D, w, 0, D, nullptr,
          bigB, (long long)D * Lp, Lp, Lp, D, D, ACT_NONE, B);
    gat_edge_kernel<<<(B * Lp + 255) / 256, 256, 0, stream>>>(bigB, asrc, adst, esrc, edst);
    gat_softmax_kernel<<<B * Lp, 256, 0, stream>>>(esrc, edst, gA, bigA);
    // x' = act(alpha @ Wh) = act(alpha @ WhT^T)  : contiguous-B fast path
    gemmN(stream, bigA, (long long)Lp * Lp, Lp, bigB, (long long)D * Lp, Lp,
          nullptr, xout, (long long)Lp * D, D, Lp, D, Lp, act, B);
  };
  gat_layer(x0, x1, w_g0, g0_asrc, g0_adst, ACT_ELU);
  gat_layer(x1, x0, w_g1, g1_asrc, g1_adst, ACT_NONE);
  _Float16* ctx = x0;

  // new_env = ctx[:,0]  (before PE is added to rows 2..)
  copy_env_kernel<<<(B * D + 255) / 256, 256, 0, stream>>>(ctx, (float*)d_out + 2 * B * D);
  add_pe_kernel<<<B * N, 256, 0, stream>>>(ctx, step, gtime);

  // ---- decoder blocks (single-query attention + FFN) ----
  auto run_dec = [&](const DecP& p, const DecW& w, const _Float16* src16,
                     const float* src32, float* out32) {
    gemmN(stream, src16, 0, D, w.wq, 0, D, p.bq, q16, 0, D, B, D, D, ACT_NONE, 1);
    gemmN(stream, ctx, 0, D, w.wk, 0, D, p.bk, bigA, 0, D, B * Lp, D, D, ACT_NONE, 1);
    gemmN(stream, ctx, 0, D, w.wv, 0, D, p.bv, bigB, 0, D, B * Lp, D, D, ACT_NONE, 1);
    dec_attn_kernel<<<B * H, 256, 0, stream>>>(q16, bigA, bigB, gmask, att16);
    gemmN(stream, att16, 0, D, w.wo, 0, D, p.bo, proj16, 0, D, B, D, D, ACT_NONE, 1);
    add_ln_kernel<<<B, 256, 0, stream>>>(src32, proj16, p.g1, p.be1, ln1_32, ln1_16);
    gemmN(stream, ln1_16, 0, D, w.w1, 0, D, p.b1, ffh, 0, DFF, B, DFF, D, ACT_RELU, 1);
    gemmN(stream, ffh, 0, DFF, w.w2, 0, DFF, p.b2, ff2, 0, D, B, D, DFF, ACT_NONE, 1);
    add_ln_kernel<<<B, 256, 0, stream>>>(ln1_32, ff2, p.g2, p.be2, out32, nullptr);
  };
  // output order: [curr_out | goal_ctx | new_env]
  run_dec(curr_p, curr_w, curr16, curr_emb, (float*)d_out);
  run_dec(goal_p, goal_w, goal16, goal_emb, (float*)d_out + B * D);
}
